// StackedJANET_64793876628235
// MI455X (gfx1250) — matrix operand order repaired and checked by reading, not compiled, not run on hardware
//
#include <hip/hip_runtime.h>
#include <hip/hip_bf16.h>
#include <stdint.h>

// ---------------------------------------------------------------------------
// Stacked JANET (2 layers, H=1024), B=64, T=512, D=512, BETA=0
// bf16 WMMA (v_wmma_f32_16x16x32_bf16) for all GEMMs, fp32 accumulation and
// fp32 elementwise recurrence. Step kernels launched per (t, layer) on the
// stream; hidden state is ping-pong double-buffered.
// ---------------------------------------------------------------------------

#define B_SZ 64
#define T_SZ 512
#define D_SZ 512
#define H_SZ 1024

typedef __attribute__((ext_vector_type(16))) __bf16 v16bf;
typedef __attribute__((ext_vector_type(8)))  float  v8f;

union FragBF {
  uint4 u[2];   // 32 bytes = 16 bf16
  v16bf v;
};

__device__ __forceinline__ unsigned short f32_to_bf16_rne(float f) {
  union { float f; unsigned int u; } c;
  c.f = f;
  unsigned int u = c.u;
  unsigned int r = u + 0x7FFFu + ((u >> 16) & 1u);  // round-to-nearest-even
  return (unsigned short)(r >> 16);
}

// ------------------------- prep kernels ------------------------------------

__global__ void cvt_f32_bf16_kernel(const float* __restrict__ src,
                                    unsigned short* __restrict__ dst, int n) {
  int i = blockIdx.x * blockDim.x + threadIdx.x;
  if (i < n) dst[i] = f32_to_bf16_rne(src[i]);
}

__global__ void bias_combine_kernel(const float* __restrict__ a,
                                    const float* __restrict__ b,
                                    float* __restrict__ dst, int n) {
  int i = blockIdx.x * blockDim.x + threadIdx.x;
  if (i < n) dst[i] = a[i] + b[i];
}

__global__ void zero_state_kernel(float* __restrict__ h0f,
                                  unsigned short* __restrict__ h0b,
                                  float* __restrict__ h1f,
                                  unsigned short* __restrict__ h1b, int n) {
  int i = blockIdx.x * blockDim.x + threadIdx.x;
  if (i < n) {
    h0f[i] = 0.0f; h1f[i] = 0.0f;
    h0b[i] = 0;    h1b[i] = 0;
  }
}

__global__ void copy_final_kernel(const float* __restrict__ h0,
                                  const float* __restrict__ h1,
                                  float* __restrict__ dst, int n) {
  int i = blockIdx.x * blockDim.x + threadIdx.x;
  if (i < n) {
    dst[i]     = h0[i];
    dst[n + i] = h1[i];
  }
}

// ------------------------- WMMA GEMM panel ---------------------------------
//
// A-fragment (16x32 bf16, contiguous k = hi*16 + s order): lane l (l16=l&15, hi=l>>4) holds
//   A[m0+l16, k0 + hi*16 .. +7] and A[m0+l16, k0+hi*16+8 .. +7]   (two b128s)
// B-fragment (32x16 bf16, same contiguous k order as A): lane l holds
//   B[k0 + hi*16 .. +15, n0+l16] == W[n0+l16, k0+hi*16 .. +15]   (two b128s,
//   contiguous along the weight row since W is stored [out, k] row-major)

__device__ __forceinline__ void mma_panel(const unsigned short* __restrict__ A,
                                          long long strideA,
                                          const unsigned short* __restrict__ Wf,
                                          const unsigned short* __restrict__ Wg,
                                          int Kw, int m0, int n0,
                                          int l16, int hi,
                                          v8f& cF, v8f& cG) {
  const unsigned short* arow  = A  + (long long)(m0 + l16) * strideA;
  const unsigned short* wfrow = Wf + (long long)(n0 + l16) * Kw + hi * 16;
  const unsigned short* wgrow = Wg + (long long)(n0 + l16) * Kw + hi * 16;
  for (int k0 = 0; k0 < Kw; k0 += 32) {
    FragBF a, bf, bg;
    a.u[0]  = *(const uint4*)(arow  + k0 + hi * 16);
    a.u[1]  = *(const uint4*)(arow  + k0 + hi * 16 + 8);
    bf.u[0] = *(const uint4*)(wfrow + k0);
    bf.u[1] = *(const uint4*)(wfrow + k0 + 8);
    bg.u[0] = *(const uint4*)(wgrow + k0);
    bg.u[1] = *(const uint4*)(wgrow + k0 + 8);
    if (k0 + 32 < Kw) {  // pull next weight K-tile toward L0/L2
      __builtin_prefetch(wfrow + k0 + 32, 0, 1);
      __builtin_prefetch(wgrow + k0 + 32, 0, 1);
    }
    // Two independent accumulator chains (F and G) -> back-to-back WMMAs
    cF = __builtin_amdgcn_wmma_f32_16x16x32_bf16(false, a.v, false, bf.v,
                                                 (short)0, cF, false, false);
    cG = __builtin_amdgcn_wmma_f32_16x16x32_bf16(false, a.v, false, bg.v,
                                                 (short)0, cG, false, false);
  }
}

// ------------------------- JANET step kernel -------------------------------
// One wave (32 threads) computes a 16(M=batch) x 16(N=hidden) tile of both
// gates, then the fused sigmoid/tanh state update.
// grid = (H/16, B/16), block = 32.

__global__ __launch_bounds__(32) void janet_step_kernel(
    const unsigned short* __restrict__ Ain, long long strideA, int Kin,
    const unsigned short* __restrict__ Hprev_bf,
    const float* __restrict__ Hprev_f32,
    const unsigned short* __restrict__ WfIn,
    const unsigned short* __restrict__ WfH,
    const unsigned short* __restrict__ WgIn,
    const unsigned short* __restrict__ WgH,
    const float* __restrict__ biasF, const float* __restrict__ biasG,
    float* __restrict__ Hnext_f32, unsigned short* __restrict__ Hnext_bf,
    float* __restrict__ OutSeq, long long outRowStride) {
  const int lane = threadIdx.x;
  const int l16  = lane & 15;
  const int hi   = lane >> 4;
  const int n0   = blockIdx.x * 16;  // hidden (output) columns
  const int m0   = blockIdx.y * 16;  // batch rows

  v8f cF = {};
  v8f cG = {};

  // input contribution: cur @ W*in^T   (K = Kin: 512 for layer0, 1024 for layer1)
  mma_panel(Ain, strideA, WfIn, WgIn, Kin, m0, n0, l16, hi, cF, cG);
  // recurrent contribution: h_prev @ W*h^T  (K = 1024)
  mma_panel(Hprev_bf, (long long)H_SZ, WfH, WgH, H_SZ, m0, n0, l16, hi, cF, cG);

  const float bF  = biasF[n0 + l16];
  const float bG  = biasG[n0 + l16];
  const int   col = n0 + l16;

  // C/D layout: VGPR r, lanes 0-15 -> M=r, lanes 16-31 -> M=r+8; N = lane&15
#pragma unroll
  for (int r = 0; r < 8; ++r) {
    const int row = m0 + r + 8 * hi;           // batch index
    const long long idx = (long long)row * H_SZ + col;
    const float F  = 1.0f / (1.0f + __expf(-(cF[r] + bF)));
    const float G  = tanhf(cG[r] + bG);
    const float hp = Hprev_f32[idx];
    const float hn = F * hp + (1.0f - F) * G;  // JANET update
    Hnext_f32[idx] = hn;
    Hnext_bf[idx]  = f32_to_bf16_rne(hn);
    if (OutSeq) OutSeq[(long long)row * outRowStride + col] = hn;
  }
}

// ------------------------- host driver -------------------------------------

extern "C" void kernel_launch(void* const* d_in, const int* in_sizes, int n_in,
                              void* d_out, int out_size, void* d_ws,
                              size_t ws_size, hipStream_t stream) {
  (void)in_sizes; (void)n_in; (void)out_size; (void)ws_size;

  // setup_inputs() order: x, then l0{Wif,bif,Whf,bhf,Wig,big,Whg,bhg}, then l1{...}
  const float* x    = (const float*)d_in[0];
  const float* W0if = (const float*)d_in[1];
  const float* b0if = (const float*)d_in[2];
  const float* W0hf = (const float*)d_in[3];
  const float* b0hf = (const float*)d_in[4];
  const float* W0ig = (const float*)d_in[5];
  const float* b0ig = (const float*)d_in[6];
  const float* W0hg = (const float*)d_in[7];
  const float* b0hg = (const float*)d_in[8];
  const float* W1if = (const float*)d_in[9];
  const float* b1if = (const float*)d_in[10];
  const float* W1hf = (const float*)d_in[11];
  const float* b1hf = (const float*)d_in[12];
  const float* W1ig = (const float*)d_in[13];
  const float* b1ig = (const float*)d_in[14];
  const float* W1hg = (const float*)d_in[15];
  const float* b1hg = (const float*)d_in[16];

  // ---- workspace carve (all chunk sizes are multiples of 16B) ----
  unsigned short* p = (unsigned short*)d_ws;
  unsigned short* x_bf = p;  p += (size_t)B_SZ * T_SZ * D_SZ;
  unsigned short* w0if = p;  p += (size_t)H_SZ * D_SZ;
  unsigned short* w0hf = p;  p += (size_t)H_SZ * H_SZ;
  unsigned short* w0ig = p;  p += (size_t)H_SZ * D_SZ;
  unsigned short* w0hg = p;  p += (size_t)H_SZ * H_SZ;
  unsigned short* w1if = p;  p += (size_t)H_SZ * H_SZ;
  unsigned short* w1hf = p;  p += (size_t)H_SZ * H_SZ;
  unsigned short* w1ig = p;  p += (size_t)H_SZ * H_SZ;
  unsigned short* w1hg = p;  p += (size_t)H_SZ * H_SZ;
  unsigned short* h0bf[2]; unsigned short* h1bf[2];
  h0bf[0] = p; p += (size_t)B_SZ * H_SZ;
  h0bf[1] = p; p += (size_t)B_SZ * H_SZ;
  h1bf[0] = p; p += (size_t)B_SZ * H_SZ;
  h1bf[1] = p; p += (size_t)B_SZ * H_SZ;
  float* fp = (float*)p;
  float* bias0F = fp; fp += H_SZ;
  float* bias0G = fp; fp += H_SZ;
  float* bias1F = fp; fp += H_SZ;
  float* bias1G = fp; fp += H_SZ;
  float* h0f[2]; float* h1f[2];
  h0f[0] = fp; fp += (size_t)B_SZ * H_SZ;
  h0f[1] = fp; fp += (size_t)B_SZ * H_SZ;
  h1f[0] = fp; fp += (size_t)B_SZ * H_SZ;
  h1f[1] = fp; fp += (size_t)B_SZ * H_SZ;

  // ---- one-time-per-call prep: bf16 conversion, bias fusion, h=0 ----
  auto cvt = [&](const float* s, unsigned short* d, int n) {
    cvt_f32_bf16_kernel<<<(n + 255) / 256, 256, 0, stream>>>(s, d, n);
  };
  cvt(x, x_bf, B_SZ * T_SZ * D_SZ);
  cvt(W0if, w0if, H_SZ * D_SZ);
  cvt(W0hf, w0hf, H_SZ * H_SZ);
  cvt(W0ig, w0ig, H_SZ * D_SZ);
  cvt(W0hg, w0hg, H_SZ * H_SZ);
  cvt(W1if, w1if, H_SZ * H_SZ);
  cvt(W1hf, w1hf, H_SZ * H_SZ);
  cvt(W1ig, w1ig, H_SZ * H_SZ);
  cvt(W1hg, w1hg, H_SZ * H_SZ);
  bias_combine_kernel<<<(H_SZ + 255) / 256, 256, 0, stream>>>(b0if, b0hf, bias0F, H_SZ);
  bias_combine_kernel<<<(H_SZ + 255) / 256, 256, 0, stream>>>(b0ig, b0hg, bias0G, H_SZ);
  bias_combine_kernel<<<(H_SZ + 255) / 256, 256, 0, stream>>>(b1if, b1hf, bias1F, H_SZ);
  bias_combine_kernel<<<(H_SZ + 255) / 256, 256, 0, stream>>>(b1ig, b1hg, bias1G, H_SZ);
  {
    int n = B_SZ * H_SZ;
    zero_state_kernel<<<(n + 255) / 256, 256, 0, stream>>>(h0f[0], h0bf[0],
                                                           h1f[0], h1bf[0], n);
  }

  // ---- sequential scan: per t, layer0 step then layer1 step ----
  float* out = (float*)d_out;  // [B, T, H] fp32, then h_stacked [2, B, H]
  dim3 grid(H_SZ / 16, B_SZ / 16);
  for (int t = 0; t < T_SZ; ++t) {
    const int pp = t & 1, nn = (t + 1) & 1;
    // layer 0: input rows are x[b, t, :], row stride T*D
    janet_step_kernel<<<grid, 32, 0, stream>>>(
        x_bf + (size_t)t * D_SZ, (long long)T_SZ * D_SZ, D_SZ,
        h0bf[pp], h0f[pp],
        w0if, w0hf, w0ig, w0hg,
        bias0F, bias0G,
        h0f[nn], h0bf[nn],
        nullptr, 0);
    // layer 1: input is layer0's freshly written h (time t); writes out[b,t,:]
    janet_step_kernel<<<grid, 32, 0, stream>>>(
        h1bf != nullptr ? h0bf[nn] : h0bf[nn], (long long)H_SZ, H_SZ,
        h1bf[pp], h1f[pp],
        w1if, w1hf, w1ig, w1hg,
        bias1F, bias1G,
        h1f[nn], h1bf[nn],
        out + (size_t)t * H_SZ, (long long)T_SZ * H_SZ);
  }

  // ---- h_stacked tail: final states live in buffer index (T % 2) == 0 ----
  {
    int n = B_SZ * H_SZ;
    copy_final_kernel<<<(n + 255) / 256, 256, 0, stream>>>(
        h0f[0], h1f[0], out + (size_t)B_SZ * T_SZ * H_SZ, n);
  }
}
